// LightningAttn3Attention_1967095021965
// MI455X (gfx1250) — matrix operand order repaired and checked by reading, not compiled, not run on hardware
//
#include <hip/hip_runtime.h>

// ---------------- problem constants ----------------
#define BB 2
#define SEQ 8192
#define DIM 1024
#define NH 16
#define NKV 4
#define HD 64
#define GROUPS (NH / NKV)
#define CHUNK 128
#define BSROWS (BB * SEQ) // 16384

typedef unsigned short ush;
typedef __bf16 bf16_t;
typedef __attribute__((ext_vector_type(16))) bf16_t v16bf;
typedef __attribute__((ext_vector_type(8)))  bf16_t v8bf;
typedef __attribute__((ext_vector_type(8)))  float  v8f;

static __device__ __forceinline__ ush h2u(bf16_t h) {
  return __builtin_bit_cast(ush, h);
}
static __device__ __forceinline__ bf16_t u2h(ush u) {
  return __builtin_bit_cast(bf16_t, u);
}
static __device__ __forceinline__ ush f2b(float f) { // native cvt path
  return h2u((bf16_t)f);
}
static __device__ __forceinline__ float b2f(ush h) {
  return __uint_as_float(((unsigned)h) << 16);
}

static __device__ __forceinline__ v8f wmma_bf16(v16bf a, v16bf b, v8f c) {
  // (neg_a, A, neg_b, B, c_mod, C, reuse_a, reuse_b)
  return __builtin_amdgcn_wmma_f32_16x16x32_bf16(false, a, false, b, (short)0, c,
                                                 false, false);
}

static __device__ __forceinline__ v16bf cat8(v8bf a, v8bf b) {
  return __builtin_shufflevector(a, b, 0, 1, 2, 3, 4, 5, 6, 7, 8, 9, 10, 11, 12,
                                 13, 14, 15);
}
// A-fragment: elements 0..7 at p[0..7], elements 8..15 at p[16..23]
static __device__ __forceinline__ v16bf ldA(const ush* p) {
  v8bf lo = *(const v8bf*)p;
  v8bf hi = *(const v8bf*)(p + 16);
  return cat8(lo, hi);
}
// B-fragment: elements 0..7 at p[0..7], elements 8..15 at p[16..23] (16-B alignment)
static __device__ __forceinline__ v16bf ldB(const ush* p) {
  v8bf lo = *(const v8bf*)p;
  v8bf hi = *(const v8bf*)(p + 16);
  return cat8(lo, hi);
}
static __device__ __forceinline__ v16bf scale16(v16bf a, float s) {
  v16bf r;
#pragma unroll
  for (int j = 0; j < 16; ++j) r[j] = (bf16_t)((float)a[j] * s);
  return r;
}

// ---------------- f32 -> bf16 convert ----------------
__global__ void cvt_f32_to_bf16(const float* __restrict__ in, ush* __restrict__ out,
                                long long n) {
  long long i = (long long)blockIdx.x * blockDim.x + threadIdx.x;
  if (i < n) out[i] = f2b(in[i]);
}

// ---------------- tiled WMMA GEMM: C[M,N] = A[M,K] * W[N,K]^T ----------------
#define BM 128
#define BN 64
#define BKK 32
#define LDA 40 // padded bf16 stride (80 B, 16-B aligned)
#define LDB 40

template <int OUT_BF16>
__global__ __launch_bounds__(256) void gemm_bf16_kernel(
    const ush* __restrict__ A, const ush* __restrict__ W, void* __restrict__ Cv,
    int M, int N, int K) {
  __shared__ __attribute__((aligned(16))) ush sA[BM * LDA];
  __shared__ __attribute__((aligned(16))) ush sB[BN * LDB];

  const int tid  = threadIdx.x;
  const int lane = tid & 31;
  const int wave = tid >> 5;
  const int wr   = wave >> 1; // 0..3 : 32-row block
  const int wc   = wave & 1;  // 0..1 : 32-col block
  const int bm   = blockIdx.y * BM;
  const int bn   = blockIdx.x * BN;

  v8f acc[2][2] = {};

  // ISA 7.12.2 fragment lane mapping (wave32)
  const int m   = lane & 15;
  const int kAo = (lane & 16) ? 8 : 0;  // A elems: K = kAo+[0..7], kAo+16+[0..7]
  const int nB  = lane & 15;
  const int kBo = (lane & 16) ? 8 : 0;  // B elems: K = kBo+[0..7], kBo+16+[0..7]
  const int mo  = (lane >> 4) * 8;      // C rows:  M = mo + vgpr

  const int ra = tid >> 1, ca = (tid & 1) * 16; // A tile staging
  const int rb = tid >> 2, cb = (tid & 3) * 8;  // B tile staging

  for (int kb = 0; kb < K; kb += BKK) {
    const uint4* ga = (const uint4*)(A + (size_t)(bm + ra) * K + kb + ca);
    uint4 a0 = ga[0];
    uint4 a1 = ga[1];
    uint4 b0 = *(const uint4*)(W + (size_t)(bn + rb) * K + kb + cb);
    *(uint4*)(sA + ra * LDA + ca)     = a0;
    *(uint4*)(sA + ra * LDA + ca + 8) = a1;
    *(uint4*)(sB + rb * LDB + cb)     = b0;
    if (kb + BKK < K) // prefetch next A slab
      __builtin_prefetch(A + (size_t)(bm + ra) * K + kb + BKK + ca, 0, 1);
    __syncthreads();

    v16bf afr[2], bfr[2];
#pragma unroll
    for (int i = 0; i < 2; ++i)
      afr[i] = ldA(sA + (size_t)(wr * 32 + i * 16 + m) * LDA + kAo);
#pragma unroll
    for (int i = 0; i < 2; ++i)
      bfr[i] = ldB(sB + (size_t)(wc * 32 + i * 16 + nB) * LDB + kBo);
#pragma unroll
    for (int i = 0; i < 2; ++i)
#pragma unroll
      for (int j = 0; j < 2; ++j) acc[i][j] = wmma_bf16(afr[i], bfr[j], acc[i][j]);
    __syncthreads();
  }

#pragma unroll
  for (int i = 0; i < 2; ++i)
#pragma unroll
    for (int j = 0; j < 2; ++j) {
      int row0 = bm + wr * 32 + i * 16 + mo;
      int col  = bn + wc * 32 + j * 16 + nB;
#pragma unroll
      for (int v = 0; v < 8; ++v) {
        if (OUT_BF16)
          ((ush*)Cv)[(size_t)(row0 + v) * N + col] = f2b(acc[i][j][v]);
        else
          ((float*)Cv)[(size_t)(row0 + v) * N + col] = acc[i][j][v];
      }
    }
}

// ---------------- RoPE (in place, bf16, layout [B,S,heads,64]) ----------------
__global__ void rope_kernel(ush* __restrict__ data, int heads, long long total) {
  long long idx = (long long)blockIdx.x * blockDim.x + threadIdx.x;
  if (idx >= total) return;
  int i        = (int)(idx & 31);
  long long rh = idx >> 5; // (b*S + s)*heads + h
  int s        = (int)((rh / heads) % SEQ);
  float inv    = __powf(10000.0f, -(float)i * (1.0f / 32.0f));
  float ang    = (float)s * inv;
  float c = __cosf(ang), sn = __sinf(ang);
  ush* p   = data + rh * HD;
  float x1 = b2f(p[i]);
  float x2 = b2f(p[i + 32]);
  p[i]      = f2b(x1 * c - x2 * sn);
  p[i + 32] = f2b(x2 * c + x1 * sn);
}

// ---------------- chunked lightning attention ----------------
// Transposed LDS tiles so every fragment read is contiguous:
//   sKT[d][t]  (64 x 128)  -> KV-update A-fragments (rows of K^T)
//   sVT[e][t]  (64 x 128)  -> V B-fragments (columns of V)
//   sKVT[e][d] (64 x 64)   -> KV B-fragments (columns of KV state)
#define LDT  136 // padded stride for 64x128 transposed tiles (272 B, 16-B mult)
#define LDKV 72  // padded stride for 64x64 transposed KV (144 B, 16-B mult)
#define LDSG 32  // per-wave score staging stride (64 B)

__global__ __launch_bounds__(256) void lightning_attn_kernel(
    const ush* __restrict__ Q, const ush* __restrict__ Km,
    const ush* __restrict__ Vm, ush* __restrict__ O) {
  __shared__ __attribute__((aligned(16))) ush sKT[HD * LDT];
  __shared__ __attribute__((aligned(16))) ush sVT[HD * LDT];
  __shared__ __attribute__((aligned(16))) ush sKVT[HD * LDKV];
  __shared__ __attribute__((aligned(16))) ush stage[8][16 * LDSG];

  const int bh = blockIdx.x; // b*NH + h
  const int b  = bh / NH;
  const int h  = bh % NH;
  const int hk = h / GROUPS;
  const float slope  = exp2f(-0.5f * (float)(h + 1));
  const float blkdec = __expf(-slope * (float)CHUNK);

  const int tid  = threadIdx.x;
  const int lane = tid & 31;
  const int wave = tid >> 5;
  const int m    = lane & 15;
  const int kAo  = (lane & 16) ? 8 : 0;
  const int nB   = lane & 15;
  const int kBo  = (lane & 16) ? 8 : 0;
  const int mo   = (lane >> 4) * 8;

  const size_t qstride = (size_t)NH * HD;  // 1024
  const size_t kstride = (size_t)NKV * HD; // 256
  const ush* Qb = Q  + (size_t)b * SEQ * qstride + (size_t)h * HD;
  const ush* Kb = Km + (size_t)b * SEQ * kstride + (size_t)hk * HD;
  const ush* Vb = Vm + (size_t)b * SEQ * kstride + (size_t)hk * HD;
  ush*       Ob = O  + (size_t)b * SEQ * qstride + (size_t)h * HD;

  v8f kvacc[4] = {}; // waves 0..3: rows [16w,16w+16) x 64 of KV state (f32)

  const float qds = __expf(-slope * (float)(16 * wave + m + 1)); // q_dec(row)

  for (int ch = 0; ch < SEQ / CHUNK; ++ch) {
    const ush* Kc = Kb + (size_t)ch * CHUNK * kstride;
    const ush* Vc = Vb + (size_t)ch * CHUNK * kstride;

    // ---- stage K/V chunk transposed into LDS (vector global read, b16 scatter)
    for (int g = tid; g < CHUNK * (HD / 8); g += 256) {
      int t = g >> 3, d0 = (g & 7) * 8;
      v8bf kv8 = *(const v8bf*)(Kc + (size_t)t * kstride + d0);
      v8bf vv8 = *(const v8bf*)(Vc + (size_t)t * kstride + d0);
#pragma unroll
      for (int j = 0; j < 8; ++j) {
        sKT[(size_t)(d0 + j) * LDT + t] = h2u(kv8[j]);
        sVT[(size_t)(d0 + j) * LDT + t] = h2u(vv8[j]);
      }
    }
    // ---- publish current KV state transposed as bf16 ----
    if (wave < 4) {
#pragma unroll
      for (int nt = 0; nt < 4; ++nt)
#pragma unroll
        for (int v = 0; v < 8; ++v)
          sKVT[(size_t)(nt * 16 + nB) * LDKV + wave * 16 + mo + v] =
              f2b(kvacc[nt][v]);
    }
    __syncthreads();

    // ---- q fragments for this wave's 16 rows (loaded once per chunk) ----
    const ush* qrow = Qb + (size_t)(ch * CHUNK + 16 * wave + m) * qstride;
    v16bf aq[2], aqs[2];
#pragma unroll
    for (int ks = 0; ks < 2; ++ks) {
      aq[ks]  = ldA(qrow + 32 * ks + kAo);
      aqs[ks] = scale16(aq[ks], qds);
    }

    v8f oacc[4] = {};

    // ---- inter-chunk: (q * q_dec) @ kv ----
#pragma unroll
    for (int ks = 0; ks < 2; ++ks)
#pragma unroll
      for (int et = 0; et < 4; ++et) {
        v16bf bb = ldB(sKVT + (size_t)(16 * et + nB) * LDKV + 32 * ks + kBo);
        oacc[et] = wmma_bf16(aqs[ks], bb, oacc[et]);
      }

    // ---- intra-chunk: decay-masked scores @ v, per 32-wide s-block ----
    ush* st = &stage[wave][0];
    const int maxsb = (16 * wave + 15) / 32; // causal skip
    for (int sb = 0; sb <= maxsb; ++sb) {
#pragma unroll
      for (int half = 0; half < 2; ++half) {
        v8f sc = {};
        const int srow = 16 * (2 * sb + half) + nB; // K row for B-frag lane
#pragma unroll
        for (int ks = 0; ks < 2; ++ks) {
          v16bf bk = ldB(Kc + (size_t)srow * kstride + 32 * ks + kBo);
          sc = wmma_bf16(aq[ks], bk, sc);
        }
#pragma unroll
        for (int v = 0; v < 8; ++v) {
          int tt = 16 * wave + mo + v;
          int sx = 32 * sb + 16 * half + nB;
          float val = (tt >= sx) ? sc[v] * __expf(-slope * (float)(tt - sx)) : 0.0f;
          st[(size_t)(mo + v) * LDSG + 16 * half + nB] = f2b(val);
        }
      }
      // (16x32 masked-score tile) @ (32x64 v-block) -- all contiguous loads
      v16bf as = ldA(st + (size_t)m * LDSG + kAo);
#pragma unroll
      for (int et = 0; et < 4; ++et) {
        v16bf bv = ldB(sVT + (size_t)(16 * et + nB) * LDT + 32 * sb + kBo);
        oacc[et] = wmma_bf16(as, bv, oacc[et]);
      }
    }

    // ---- write O chunk (bf16, layout [B,S,NH,HD]) ----
#pragma unroll
    for (int et = 0; et < 4; ++et)
#pragma unroll
      for (int v = 0; v < 8; ++v) {
        int tt = 16 * wave + mo + v;
        Ob[(size_t)(ch * CHUNK + tt) * qstride + 16 * et + nB] = f2b(oacc[et][v]);
      }
    __syncthreads();

    // ---- KV update: kv = blk_dec*kv + (k*k_dec)^T @ v (waves 0..3) ----
    if (wave < 4) {
#pragma unroll
      for (int nt = 0; nt < 4; ++nt)
#pragma unroll
        for (int v = 0; v < 8; ++v) kvacc[nt][v] *= blkdec;
      const int d = 16 * wave + m; // A row = d (contiguous row of sKT)
#pragma unroll
      for (int kk = 0; kk < 4; ++kk) {
        const int tb = 32 * kk + kAo;
        v16bf kraw = ldA(sKT + (size_t)d * LDT + tb);
        v16bf ak;
#pragma unroll
        for (int j = 0; j < 16; ++j) {
          int t = tb + ((j < 8) ? j : (8 + j));
          ak[j] = (bf16_t)((float)kraw[j] *
                           __expf(-slope * (float)(CHUNK - 1 - t)));
        }
#pragma unroll
        for (int et = 0; et < 4; ++et) {
          v16bf bv = ldB(sVT + (size_t)(16 * et + nB) * LDT + 32 * kk + kBo);
          kvacc[et] = wmma_bf16(ak, bv, kvacc[et]);
        }
      }
    }
    __syncthreads();
  }
}

// ---------------- host launch ----------------
extern "C" void kernel_launch(void* const* d_in, const int* in_sizes, int n_in,
                              void* d_out, int out_size, void* d_ws, size_t ws_size,
                              hipStream_t stream) {
  (void)in_sizes; (void)n_in; (void)out_size; (void)ws_size;
  const float* x  = (const float*)d_in[0];
  const float* Wq = (const float*)d_in[1];
  const float* Wk = (const float*)d_in[2];
  const float* Wv = (const float*)d_in[3];
  const float* Wo = (const float*)d_in[4];

  ush* ws = (ush*)d_ws;
  size_t off = 0;
  ush* xbf = ws + off; off += (size_t)BSROWS * DIM;
  ush* wqb = ws + off; off += (size_t)(NH * HD) * DIM;
  ush* wkb = ws + off; off += (size_t)(NKV * HD) * DIM;
  ush* wvb = ws + off; off += (size_t)(NKV * HD) * DIM;
  ush* wob = ws + off; off += (size_t)DIM * (NH * HD);
  ush* qbf = ws + off; off += (size_t)BSROWS * (NH * HD);
  ush* kbf = ws + off; off += (size_t)BSROWS * (NKV * HD);
  ush* vbf = ws + off; off += (size_t)BSROWS * (NKV * HD);
  ush* obf = ws + off; off += (size_t)BSROWS * (NH * HD);

  {
    long long n;
    n = (long long)BSROWS * DIM;
    cvt_f32_to_bf16<<<(int)((n + 255) / 256), 256, 0, stream>>>(x, xbf, n);
    n = (long long)(NH * HD) * DIM;
    cvt_f32_to_bf16<<<(int)((n + 255) / 256), 256, 0, stream>>>(Wq, wqb, n);
    n = (long long)(NKV * HD) * DIM;
    cvt_f32_to_bf16<<<(int)((n + 255) / 256), 256, 0, stream>>>(Wk, wkb, n);
    cvt_f32_to_bf16<<<(int)((n + 255) / 256), 256, 0, stream>>>(Wv, wvb, n);
    n = (long long)DIM * (NH * HD);
    cvt_f32_to_bf16<<<(int)((n + 255) / 256), 256, 0, stream>>>(Wo, wob, n);
  }

  // projections: q/k/v = x @ W^T  (bf16 out)
  gemm_bf16_kernel<1><<<dim3((NH * HD) / BN, BSROWS / BM), 256, 0, stream>>>(
      xbf, wqb, qbf, BSROWS, NH * HD, DIM);
  gemm_bf16_kernel<1><<<dim3((NKV * HD) / BN, BSROWS / BM), 256, 0, stream>>>(
      xbf, wkb, kbf, BSROWS, NKV * HD, DIM);
  gemm_bf16_kernel<1><<<dim3((NKV * HD) / BN, BSROWS / BM), 256, 0, stream>>>(
      xbf, wvb, vbf, BSROWS, NKV * HD, DIM);

  // RoPE (in place)
  long long qtot = (long long)BSROWS * NH * 32;
  rope_kernel<<<(int)((qtot + 255) / 256), 256, 0, stream>>>(qbf, NH, qtot);
  long long ktot = (long long)BSROWS * NKV * 32;
  rope_kernel<<<(int)((ktot + 255) / 256), 256, 0, stream>>>(kbf, NKV, ktot);

  // chunked lightning attention, one block per (b, h)
  lightning_attn_kernel<<<BB * NH, 256, 0, stream>>>(qbf, kbf, vbf, obf);

  // output projection: out = O @ Wo^T (f32 out)
  gemm_bf16_kernel<0><<<dim3(DIM / BN, BSROWS / BM), 256, 0, stream>>>(
      obf, wob, d_out, BSROWS, DIM, DIM);
}
